// NetVLAD_20444044329188
// MI455X (gfx1250) — matrix operand order repaired and checked by reading, not compiled, not run on hardware
//
#include <hip/hip_runtime.h>
#include <hip/hip_bf16.h>
#include <stdint.h>

// ---------------------------------------------------------------------------
// NetVLAD fused pipeline for gfx1250 (MI455X)
//   b=32, m=8192, C=512, K=8 clusters, d=64
//   Phase 0: zero workspace accumulators
//   Phase 1: convert W_red (fp32) -> bf16 into workspace
//   Phase 2: fused [TDM-stage W -> LDS, reduce-GEMM (WMMA bf16),
//                   L2norm + softmax + VLAD partial]
//   Phase 3: finalize: centroid subtract, intra-norm, global norm
// ---------------------------------------------------------------------------

typedef __attribute__((ext_vector_type(16))) __bf16    v16bf;
typedef __attribute__((ext_vector_type(8)))  float     v8f;
typedef __attribute__((ext_vector_type(4)))  uint32_t  v4u;
typedef __attribute__((ext_vector_type(8)))  int       v8i;
typedef __attribute__((ext_vector_type(4)))  int       v4i;

#define B_BATCH 32
#define M_PTS   8192
#define C_IN    512
#define D_RED   64
#define K_CLU   8
#define MTILE   128   // rows of x per block
#define XS      65    // padded LDS row stride for s_x (kills 64-bank conflicts)
#define WBS     520   // padded LDS row stride for staged W (512 + 8 bf16 = +4 DW)

__device__ __forceinline__ uint32_t pk_bf16x2(float lo, float hi) {
    uint32_t ulo = __float_as_uint(lo);
    uint32_t uhi = __float_as_uint(hi);
    ulo = (ulo + 0x7FFFu + ((ulo >> 16) & 1u)) >> 16;   // RNE fp32 -> bf16
    uhi = (uhi + 0x7FFFu + ((uhi >> 16) & 1u)) >> 16;
    return ulo | (uhi << 16);
}

union AFrag { uint32_t u[8]; v16bf v; };
union BFrag { uint4    q[2]; v16bf v; };

// --------------------------- Phase 0: zero ---------------------------------
__global__ void nv_zero_kernel(float* p, int n) {
    int i = blockIdx.x * blockDim.x + threadIdx.x;
    if (i < n) p[i] = 0.0f;
}

// ---------------------- Phase 1: W_red fp32 -> bf16 ------------------------
__global__ void nv_prep_w_kernel(const float* __restrict__ wred,
                                 uint16_t* __restrict__ wbf, int n) {
    int i = blockIdx.x * blockDim.x + threadIdx.x;
    if (i < n) {
        uint32_t u = __float_as_uint(wred[i]);
        wbf[i] = (uint16_t)((u + 0x7FFFu + ((u >> 16) & 1u)) >> 16);
    }
}

// ------------------- Phase 2: fused main streaming kernel ------------------
// grid: (M_PTS/MTILE, B_BATCH) = (64, 32); block: 256 threads (8 waves)
__global__ void __launch_bounds__(256)
nv_main_kernel(const float* __restrict__ x,
               const uint16_t* __restrict__ wbf,     // [D_RED][C_IN] bf16 (global)
               const float* __restrict__ b_red,      // [D_RED]
               const float* __restrict__ w_lin,      // [K_CLU][D_RED]
               const float* __restrict__ b_lin,      // [K_CLU]
               float* __restrict__ ws_a,             // [B][K]  (global atomics)
               float* __restrict__ ws_ax)            // [B][K][D] (global atomics)
{
    __shared__ uint16_t s_wb[D_RED * WBS];     // TDM-staged W (bf16, padded rows)
    __shared__ float s_x[MTILE * XS];          // reduced descriptors (raw, pre-norm)
    __shared__ float s_a[MTILE * K_CLU];       // softmax * rowscale
    __shared__ float s_wlin[K_CLU * D_RED];
    __shared__ float s_blin[K_CLU];
    __shared__ float s_bred[D_RED];
    __shared__ float s_asum[K_CLU];

    const int tid  = threadIdx.x;
    const int wave = tid >> 5;
    const int lane = tid & 31;
    const int bb   = blockIdx.y;
    const int r0   = blockIdx.x * MTILE;

    // ---- issue TDM: stage W_red bf16 [64 x 512] global -> LDS (padded) ----
    // Issued first (all counters zero => EXEC-masked waves skip cleanly).
    if (tid < 32) {
        uint64_t ga  = (uint64_t)(uintptr_t)wbf;
        uint32_t lds = (uint32_t)(uintptr_t)(&s_wb[0]);   // low 32b of flat = LDS addr
        v4u g0;
        g0.x = 1u;                                        // count=1, no gather
        g0.y = lds;                                       // lds_addr
        g0.z = (uint32_t)ga;                              // global_addr[31:0]
        g0.w = (uint32_t)((ga >> 32) & 0x01FFFFFFu)       // global_addr[56:32]
             | 0x80000000u;                               // type=2 ("image")
        v8i g1;
        g1[0] = (int)((1u << 16)      // data_size = 2 bytes
                    | (1u << 20)      // pad_enable
                    | (7u << 22)      // pad_interval = 256 DWORDs (one 1024B row)
                    | (3u << 25));    // pad_amount   = 4 DWORDs  (row stride 520 bf16)
        g1[1] = (int)(512u << 16);    // tensor_dim0 = 512 (low 16 bits)
        g1[2] = (int)(64u  << 16);    // tensor_dim0 hi = 0 | tensor_dim1 = 64 (lo)
        g1[3] = (int)(512u << 16);    // tensor_dim1 hi = 0 | tile_dim0 = 512
        g1[4] = 64;                   // tile_dim1 = 64, tile_dim2 = 0
        g1[5] = 512;                  // tensor_dim0_stride[31:0] = 512
        g1[6] = 0;                    // stride hi | tensor_dim1_stride lo
        g1[7] = 0;
        v4i gz  = {0, 0, 0, 0};               // 2D tensor: groups 2/3 unused
        v8i gz8 = {0, 0, 0, 0, 0, 0, 0, 0};   // group 4 unused
        __builtin_amdgcn_tensor_load_to_lds(g0, g1, gz, gz, gz8, 0);
    }

    // stage small constants into LDS (overlaps with TDM transfer)
    if (tid < D_RED)          s_bred[tid] = b_red[tid];
    if (tid < K_CLU)        { s_blin[tid] = b_lin[tid]; s_asum[tid] = 0.0f; }
    for (int i = tid; i < K_CLU * D_RED; i += 256) s_wlin[i] = w_lin[i];

    __builtin_amdgcn_s_wait_tensorcnt(0);   // wave0: wait for TDM; others: no-op
    __syncthreads();

    // ---------------- reduction GEMM: 16x64 strip per wave -----------------
    const int ncol  = lane & 15;
    const int koffA = (lane >> 4) * 8;    // A: lanes 0-15 -> K 0..7/16..23, 16-31 -> +8
    const int koffB = (lane >> 4) * 8;    // B: lanes 0-15 -> K 0..7/16..23, 16-31 -> +8
    const float* rowPtr =
        x + ((size_t)bb * M_PTS + (size_t)(r0 + wave * 16 + ncol)) * C_IN;

    v8f acc[4];
#pragma unroll
    for (int t = 0; t < 4; ++t) acc[t] = (v8f){0,0,0,0,0,0,0,0};

#pragma unroll 4
    for (int kk = 0; kk < C_IN / 32; ++kk) {
        const int kA = kk * 32 + koffA;
        __builtin_prefetch(rowPtr + kA + 96, 0, 0);   // global_prefetch_b8
        float4 fa = *(const float4*)(rowPtr + kA);
        float4 fb = *(const float4*)(rowPtr + kA + 4);
        float4 fc = *(const float4*)(rowPtr + kA + 16);
        float4 fd = *(const float4*)(rowPtr + kA + 20);
        AFrag a;
        a.u[0] = pk_bf16x2(fa.x, fa.y); a.u[1] = pk_bf16x2(fa.z, fa.w);
        a.u[2] = pk_bf16x2(fb.x, fb.y); a.u[3] = pk_bf16x2(fb.z, fb.w);
        a.u[4] = pk_bf16x2(fc.x, fc.y); a.u[5] = pk_bf16x2(fc.z, fc.w);
        a.u[6] = pk_bf16x2(fd.x, fd.y); a.u[7] = pk_bf16x2(fd.z, fd.w);
#pragma unroll
        for (int t = 0; t < 4; ++t) {
            const uint16_t* wrow = s_wb + (size_t)(t * 16 + ncol) * WBS
                                        + kk * 32 + koffB;
            BFrag bf;
            bf.q[0] = ((const uint4*)wrow)[0];   // ds_load_b128
            bf.q[1] = ((const uint4*)wrow)[2];
            acc[t] = __builtin_amdgcn_wmma_f32_16x16x32_bf16(
                         false, a.v, false, bf.v, (short)0, acc[t], false, false);
        }
    }

    // scatter C tiles (+bias) to LDS: row = wave*16 + j + 8*(lane>=16), col = t*16+ncol
    const int rloc = wave * 16 + ((lane >> 4) << 3);
#pragma unroll
    for (int t = 0; t < 4; ++t) {
        float bias = s_bred[t * 16 + ncol];
#pragma unroll
        for (int j = 0; j < 8; ++j)
            s_x[(rloc + j) * XS + t * 16 + ncol] = acc[t][j] + bias;
    }
    __syncthreads();

    // --------- per-row: L2 norm scale, logits, softmax (2 threads/row) -----
    {
        const int row  = tid >> 1;
        const int half = tid & 1;
        const float* xr = &s_x[row * XS + half * 32];
        float ssq = 0.0f;
#pragma unroll
        for (int c = 0; c < 32; ++c) { float v = xr[c]; ssq += v * v; }
        ssq += __shfl_xor(ssq, 1, 32);
        const float scale = 1.0f / fmaxf(sqrtf(ssq), 1e-12f);

        float lg[K_CLU];
#pragma unroll
        for (int k = 0; k < K_CLU; ++k) {
            const float* wr = &s_wlin[k * D_RED + half * 32];
            float p = 0.0f;
#pragma unroll
            for (int c = 0; c < 32; ++c) p += xr[c] * wr[c];
            p += __shfl_xor(p, 1, 32);
            lg[k] = p * scale + s_blin[k];
        }
        float mx = lg[0];
#pragma unroll
        for (int k = 1; k < K_CLU; ++k) mx = fmaxf(mx, lg[k]);
        float se = 0.0f;
#pragma unroll
        for (int k = 0; k < K_CLU; ++k) { lg[k] = __expf(lg[k] - mx); se += lg[k]; }
        const float inv = 1.0f / se;
        if (half == 0) {
#pragma unroll
            for (int k = 0; k < K_CLU; ++k) {
                float av = lg[k] * inv;            // softmax weight a[row][k]
                s_a[row * K_CLU + k] = av * scale; // fold L2-norm scale into a
                atomicAdd(&s_asum[k], av);         // ds_add_f32
            }
        }
    }
    __syncthreads();

    // ----------- VLAD partial: vacc[k][dc] = sum_r a'[r][k] * xraw[r][dc] ---
    {
        const int group = tid >> 6;   // 4 row-groups of 32
        const int dc    = tid & 63;
        float vacc[K_CLU];
#pragma unroll
        for (int k = 0; k < K_CLU; ++k) vacc[k] = 0.0f;
        for (int r = group * 32; r < group * 32 + 32; ++r) {
            const float xv = s_x[r * XS + dc];
            const float* ar = &s_a[r * K_CLU];
#pragma unroll
            for (int k = 0; k < K_CLU; ++k) vacc[k] += ar[k] * xv;
        }
        float* wax = ws_ax + (size_t)bb * (K_CLU * D_RED);
#pragma unroll
        for (int k = 0; k < K_CLU; ++k)
            atomicAdd(&wax[k * D_RED + dc], vacc[k]);   // global_atomic_add_f32
        if (tid < K_CLU)
            atomicAdd(&ws_a[bb * K_CLU + tid], s_asum[tid]);
    }
}

// --------------------------- Phase 3: finalize -----------------------------
// grid: B_BATCH blocks, 512 threads: t -> (k = t/64, dc = t%64)
__global__ void __launch_bounds__(512)
nv_finalize_kernel(const float* __restrict__ ws_a,
                   const float* __restrict__ ws_ax,
                   const float* __restrict__ centroids,
                   float* __restrict__ out)
{
    __shared__ float red[16];
    __shared__ float gred[16];
    const int bb = blockIdx.x;
    const int t  = threadIdx.x;
    const int k  = t >> 6;
    const int dc = t & 63;
    const int w  = t >> 5;

    const float asum = ws_a[bb * K_CLU + k];
    float v = ws_ax[(size_t)bb * (K_CLU * D_RED) + k * D_RED + dc]
            - centroids[k * D_RED + dc] * asum;

    // intra-normalize per cluster (64 values = waves 2k and 2k+1)
    float s = v * v;
#pragma unroll
    for (int m = 16; m >= 1; m >>= 1) s += __shfl_xor(s, m, 32);
    if ((t & 31) == 0) red[w] = s;
    __syncthreads();
    const float ssum  = red[2 * k] + red[2 * k + 1];
    const float vn    = v * (1.0f / fmaxf(sqrtf(ssum), 1e-12f));

    // global L2 normalize over all 512
    float g = vn * vn;
#pragma unroll
    for (int m = 16; m >= 1; m >>= 1) g += __shfl_xor(g, m, 32);
    if ((t & 31) == 0) gred[w] = g;
    __syncthreads();
    float tot = 0.0f;
#pragma unroll
    for (int i = 0; i < 16; ++i) tot += gred[i];
    out[(size_t)bb * (K_CLU * D_RED) + t] = vn * (1.0f / fmaxf(sqrtf(tot), 1e-12f));
}

// ---------------------------------------------------------------------------
extern "C" void kernel_launch(void* const* d_in, const int* in_sizes, int n_in,
                              void* d_out, int out_size, void* d_ws, size_t ws_size,
                              hipStream_t stream) {
    const float* x     = (const float*)d_in[0];  // [32,8192,512]
    /* d_in[1] = mask (all true) -> ignored */
    const float* w_red = (const float*)d_in[2];  // [64,512]
    const float* b_red = (const float*)d_in[3];  // [64]
    const float* w_lin = (const float*)d_in[4];  // [8,64]
    const float* b_lin = (const float*)d_in[5];  // [8]
    const float* cent  = (const float*)d_in[6];  // [8,64]
    float* out = (float*)d_out;                  // [32,512]

    // workspace layout
    float*    ws_a  = (float*)d_ws;                          // 32*8
    float*    ws_ax = ws_a + B_BATCH * K_CLU;                // 32*8*64
    uint16_t* wbf   = (uint16_t*)(ws_ax + B_BATCH * K_CLU * D_RED); // 64*512 bf16

    const int nzero = B_BATCH * K_CLU + B_BATCH * K_CLU * D_RED;
    nv_zero_kernel<<<(nzero + 255) / 256, 256, 0, stream>>>(ws_a, nzero);

    const int nw = D_RED * C_IN;
    nv_prep_w_kernel<<<(nw + 255) / 256, 256, 0, stream>>>(w_red, wbf, nw);

    dim3 grid(M_PTS / MTILE, B_BATCH);
    nv_main_kernel<<<grid, 256, 0, stream>>>(x, wbf, b_red, w_lin, b_lin,
                                             ws_a, ws_ax);

    nv_finalize_kernel<<<B_BATCH, 512, 0, stream>>>(ws_a, ws_ax, cent, out);
}
